// POS_GAT_Layer_21053929685022
// MI455X (gfx1250) — matrix operand order repaired and checked by reading, not compiled, not run on hardware
//
#include <hip/hip_runtime.h>

#define N_NODES 80000
#define N_EDGES 1280000
#define IN_DIM 512
#define OUT_DIM 64
#define POS_DIM 8
#define NEG_SLOPE 0.01f

typedef __attribute__((ext_vector_type(16))) __bf16 v16bf;
typedef __attribute__((ext_vector_type(8)))  float v8f;

// ---------- workspace layout (bytes) ----------
#define WS_WBF    ((size_t)0)                      // 64*512 bf16      = 65536
#define WS_Z      ((size_t)65536)                  // 80000*64 f32     = 20480000
#define WS_EL     (WS_Z   + (size_t)20480000)      // 80000 f32
#define WS_ER     (WS_EL  + (size_t)320000)
#define WS_EMAX   (WS_ER  + (size_t)320000)        // 80000 u32 (ordered-float keys)
#define WS_DENOM  (WS_EMAX+ (size_t)320000)        // 80000 f32
#define WS_EX     (WS_DENOM+(size_t)320000)        // 1280000 f32 (e, then ex)

// monotonic float <-> uint mapping so atomicMax(u32) == float max
__device__ __forceinline__ unsigned f2ord(float f) {
    unsigned u = __float_as_uint(f);
    return (u >> 31) ? ~u : (u | 0x80000000u);
}
__device__ __forceinline__ float ord2f(unsigned k) {
    unsigned u = (k >> 31) ? (k & 0x7FFFFFFFu) : ~k;
    return __uint_as_float(u);
}

// ---------- K0: convert W_fc (64x512 f32, row-major) to bf16 ----------
__global__ void k_cvt_w(const float* __restrict__ W, __bf16* __restrict__ Wb) {
    int i = blockIdx.x * blockDim.x + threadIdx.x;
    if (i < OUT_DIM * IN_DIM) Wb[i] = (__bf16)W[i];
}

// ---------- K1: init accumulators (harness poisons buffers) ----------
__global__ void k_init(float* __restrict__ hout, float* __restrict__ denom,
                       unsigned* __restrict__ emaxk) {
    int i = blockIdx.x * blockDim.x + threadIdx.x;
    if (i < N_NODES * OUT_DIM) hout[i] = 0.0f;
    if (i < N_NODES) { denom[i] = 0.0f; emaxk[i] = 0u; }  // key 0 == -inf sentinel
}

// ---------- K2: z = h @ W^T via v_wmma_f32_16x16x32_bf16 ----------
// One wave -> 16 rows x 64 cols. 8 waves/block, 625 blocks (exact, EXEC always full).
// kc loop fully unrolled: the scheduler pipelines loads across chunks itself.
__global__ void __launch_bounds__(256) k_gemm(const float* __restrict__ h,
                                              const __bf16* __restrict__ Wb,
                                              float* __restrict__ z) {
    const int lane = threadIdx.x & 31;
    const int wave = threadIdx.x >> 5;
    const int rowBase = (blockIdx.x * 8 + wave) * 16;
    const int mlo  = lane & 15;     // row within slab (A) / column within tile (B,D)
    const int half = lane >> 4;     // K-half selector

    const float*  __restrict__ hrow  = h  + (size_t)(rowBase + mlo) * IN_DIM;
    const __bf16* __restrict__ wbase = Wb + (size_t)mlo * IN_DIM + 16 * half;

    v8f acc[4] = {v8f{}, v8f{}, v8f{}, v8f{}};

    #pragma unroll
    for (int kc = 0; kc < IN_DIM; kc += 32) {
        // A-matrix lane fragment: 16 contiguous bf16 at K = kc + 16*half + {0..15}
        union { v16bf v; __bf16 e[16]; } a;
        #pragma unroll
        for (int vr = 0; vr < 8; ++vr) {
            int kb = kc + 16 * half + 2 * vr;
            float2 hv = *(const float2*)(hrow + kb);
            a.e[2 * vr]     = (__bf16)hv.x;   // -> v_cvt_pk_bf16_f32
            a.e[2 * vr + 1] = (__bf16)hv.y;
        }
        // B-matrix lane fragment: 16 contiguous bf16 of W row n, K = kc + 16*half
        #pragma unroll
        for (int t = 0; t < 4; ++t) {
            v16bf b = *(const v16bf*)(wbase + (size_t)t * 16 * IN_DIM + kc);
            acc[t] = __builtin_amdgcn_wmma_f32_16x16x32_bf16(
                false, a.v, false, b, (short)0, acc[t], false, false);
        }
    }

    // D layout: VGPR r -> row rowBase + r + 8*half, col = t*16 + mlo (coalesced 64B/half-wave)
    #pragma unroll
    for (int r = 0; r < 8; ++r) {
        float* zp = z + (size_t)(rowBase + r + 8 * half) * OUT_DIM + mlo;
        zp[0]  = acc[0][r];
        zp[16] = acc[1][r];
        zp[32] = acc[2][r];
        zp[48] = acc[3][r];
    }
}

// ---------- K3: el/er per node ----------
__global__ void k_att(const float* __restrict__ z, const float* __restrict__ pos,
                      const float* __restrict__ Wa,
                      float* __restrict__ el, float* __restrict__ er) {
    int i = blockIdx.x * blockDim.x + threadIdx.x;
    if (i >= N_NODES) return;
    const float* zr = z + (size_t)i * OUT_DIM;
    float a = 0.0f, b = 0.0f;
    #pragma unroll
    for (int d = 0; d < OUT_DIM; d += 4) {
        float4 zv = *(const float4*)(zr + d);
        float4 ws = *(const float4*)(Wa + d);
        float4 wd = *(const float4*)(Wa + OUT_DIM + d);
        a += zv.x * ws.x + zv.y * ws.y + zv.z * ws.z + zv.w * ws.w;
        b += zv.x * wd.x + zv.y * wd.y + zv.z * wd.z + zv.w * wd.w;
    }
    const float* pr = pos + (size_t)i * POS_DIM;
    #pragma unroll
    for (int j = 0; j < POS_DIM; ++j) {
        float p = pr[j];
        a += p * Wa[2 * OUT_DIM + j];
        b += p * Wa[2 * OUT_DIM + POS_DIM + j];
    }
    el[i] = a;
    er[i] = b;
}

// ---------- K4: e = leakyrelu(el[src]+er[dst]); segment max via u32 atomicMax ----------
__global__ void k_edge_max(const int* __restrict__ src, const int* __restrict__ dst,
                           const float* __restrict__ el, const float* __restrict__ er,
                           float* __restrict__ e, unsigned* __restrict__ emaxk) {
    int i = blockIdx.x * blockDim.x + threadIdx.x;
    if (i >= N_EDGES) return;
    int d = dst[i];
    float v = el[src[i]] + er[d];
    v = (v > 0.0f) ? v : NEG_SLOPE * v;
    e[i] = v;
    atomicMax(emaxk + d, f2ord(v));
}

// ---------- K5: ex = exp(e - emax[dst]); denom[dst] += ex ----------
__global__ void k_edge_exp(const int* __restrict__ dst, const unsigned* __restrict__ emaxk,
                           float* __restrict__ e_ex, float* __restrict__ denom) {
    int i = blockIdx.x * blockDim.x + threadIdx.x;
    if (i >= N_EDGES) return;
    int d = dst[i];
    float v = __expf(e_ex[i] - ord2f(emaxk[d]));
    e_ex[i] = v;
    atomicAdd(denom + d, v);
}

// ---------- K6: h_out[dst] += (ex/denom[dst]) * z[src]; 64 lanes per edge ----------
__global__ void __launch_bounds__(256) k_scatter(const int* __restrict__ src,
                                                 const int* __restrict__ dst,
                                                 const float* __restrict__ ex,
                                                 const float* __restrict__ denom,
                                                 const float* __restrict__ z,
                                                 float* __restrict__ hout) {
    long long t = (long long)blockIdx.x * 256 + threadIdx.x;
    int edge = (int)(t >> 6);
    int dim  = (int)(t & 63);
    if (edge >= N_EDGES) return;
    int s = src[edge], d = dst[edge];
    float alpha = ex[edge] / denom[d];
    atomicAdd(hout + (size_t)d * OUT_DIM + dim, alpha * z[(size_t)s * OUT_DIM + dim]);
}

extern "C" void kernel_launch(void* const* d_in, const int* in_sizes, int n_in,
                              void* d_out, int out_size, void* d_ws, size_t ws_size,
                              hipStream_t stream) {
    const float* h    = (const float*)d_in[0];
    const float* pos  = (const float*)d_in[1];
    const int*   src  = (const int*)d_in[2];
    const int*   dst  = (const int*)d_in[3];
    const float* Wfc  = (const float*)d_in[4];
    const float* Wat  = (const float*)d_in[5];
    float* hout = (float*)d_out;

    char* ws = (char*)d_ws;
    __bf16*   Wb    = (__bf16*)(ws + WS_WBF);
    float*    z     = (float*)(ws + WS_Z);
    float*    el    = (float*)(ws + WS_EL);
    float*    er    = (float*)(ws + WS_ER);
    unsigned* emaxk = (unsigned*)(ws + WS_EMAX);
    float*    denom = (float*)(ws + WS_DENOM);
    float*    e_ex  = (float*)(ws + WS_EX);

    k_cvt_w<<<(OUT_DIM * IN_DIM + 255) / 256, 256, 0, stream>>>(Wfc, Wb);
    k_init<<<(N_NODES * OUT_DIM + 255) / 256, 256, 0, stream>>>(hout, denom, emaxk);
    k_gemm<<<N_NODES / (16 * 8), 256, 0, stream>>>(h, Wb, z);           // 625 blocks
    k_att<<<(N_NODES + 255) / 256, 256, 0, stream>>>(z, pos, Wat, el, er);
    k_edge_max<<<(N_EDGES + 255) / 256, 256, 0, stream>>>(src, dst, el, er, e_ex, emaxk);
    k_edge_exp<<<(N_EDGES + 255) / 256, 256, 0, stream>>>(dst, emaxk, e_ex, denom);
    k_scatter<<<(int)(((long long)N_EDGES * 64 + 255) / 256), 256, 0, stream>>>(
        src, dst, e_ex, denom, z, hout);
}
